// SliceLSTM_59442347377082
// MI455X (gfx1250) — matrix operand order repaired and checked by reading, not compiled, not run on hardware
//
#include <hip/hip_runtime.h>
#include <hip/hip_bf16.h>

// ---------------------------------------------------------------------------
// SliceLSTM for MI455X (gfx1250): bf16 WMMA path, weights resident in L2.
// ---------------------------------------------------------------------------

typedef __attribute__((ext_vector_type(16))) __bf16 bf16x16;
typedef __attribute__((ext_vector_type(8)))  __bf16 bf16x8;
typedef __attribute__((ext_vector_type(8)))  float  v8f;

#define BS   64
#define TT   512
#define NS   4
#define IN_S 256
#define H_S  256
#define HID  1024   // NS*H_S
#define FEAT 1024   // NS*IN_S

__device__ __forceinline__ v8f wmma_bf16(bf16x16 a, bf16x16 b, v8f c) {
    return __builtin_amdgcn_wmma_f32_16x16x32_bf16(
        /*neg_a=*/false, a, /*neg_b=*/false, b,
        /*c_mod=*/(short)0, c, /*reuse_a=*/false, /*reuse_b=*/false);
}

__device__ __forceinline__ float sigf(float x) { return 1.0f / (1.0f + __expf(-x)); }

// Build 16x32 bf16 A-fragment from row-major fp32 activations.
// Lane holds row m=l%16; K-chunks at kbase+(l/16)*8 and +16 (2 bf16 / VGPR).
__device__ __forceinline__ bf16x16 load_a_f32(const float* __restrict__ p) {
    bf16x16 a;
#pragma unroll
    for (int i = 0; i < 8; ++i) a[i] = (__bf16)p[i];
#pragma unroll
    for (int i = 0; i < 8; ++i) a[8 + i] = (__bf16)p[16 + i];
    return a;
}

__device__ __forceinline__ bf16x16 load_a_bf16(const __bf16* __restrict__ p) {
    bf16x8 lo = *(const bf16x8*)p;
    bf16x8 hi = *(const bf16x8*)(p + 16);
    bf16x16 a;
#pragma unroll
    for (int i = 0; i < 8; ++i) { a[i] = lo[i]; a[8 + i] = hi[i]; }
    return a;
}

// ---------------------------------------------------------------------------
// One-time prep: fp32 [S][K][N] -> bf16 [S][N][K]  (B operand wants [N,K])
// ---------------------------------------------------------------------------
__global__ void k_transpose_bf16(const float* __restrict__ src,
                                 __bf16* __restrict__ dst,
                                 int Sb, int K, int N) {
    size_t i = (size_t)blockIdx.x * blockDim.x + threadIdx.x;
    size_t total = (size_t)Sb * K * N;
    if (i >= total) return;
    int n = (int)(i % N);
    size_t r = i / N;
    int k = (int)(r % K);
    int s = (int)(r / K);
    dst[((size_t)s * N + n) * K + k] = (__bf16)src[i];
}

__global__ void k_init_state(__bf16* __restrict__ hbf, float* __restrict__ cst) {
    int i = blockIdx.x * blockDim.x + threadIdx.x;   // 65536 threads
    hbf[i] = (__bf16)0.0f;
    cst[i] = 0.0f;
}

// ---------------------------------------------------------------------------
// Kernel A: per-slice gate GEMMs + activations.
// gates[b, s, j] = sum_k x_t[b,s,k] Ws[s,k,j] + sum_k h[b,s,k] Us[s,k,j] + bias
// 1024 waves: wave -> (s, mt, nt); tile = rows mt*16..+15 (batch), cols nt*16..+15
// Output: activated gates as bf16, gate-major: Gbf[g][b*HID + s*256 + jj]
// ---------------------------------------------------------------------------
__global__ void __launch_bounds__(128)
k_gates(const float* __restrict__ x,
        const __bf16* __restrict__ Wt,      // [S][1024][256] bf16
        const __bf16* __restrict__ Ut,      // [S][1024][256] bf16
        const __bf16* __restrict__ hbf,     // [64][1024] bf16
        const float* __restrict__ biases,   // [S][1024]
        __bf16* __restrict__ Gbf,           // [4][64][1024] bf16
        int t) {
    int wave = (blockIdx.x * blockDim.x + threadIdx.x) >> 5;   // 0..1023
    int lane = threadIdx.x & 31;
    int s  = wave >> 8;         // slice 0..3
    int rm = wave & 255;
    int mt = rm >> 6;           // 0..3  batch tile
    int nt = rm & 63;           // 0..63 column tile
    int hi = lane >> 4;
    int lo = lane & 15;

    const float*  xrow = x   + ((size_t)(mt * 16 + lo) * TT + t) * FEAT + s * IN_S;
    const __bf16* hrow = hbf + (size_t)(mt * 16 + lo) * HID + s * H_S;
    const __bf16* wB   = Wt + ((size_t)s * 1024 + nt * 16 + lo) * 256;
    const __bf16* uB   = Ut + ((size_t)s * 1024 + nt * 16 + lo) * 256;

    v8f acc = {};
#pragma unroll
    for (int kk = 0; kk < 256; kk += 32) {                    // x @ Ws
        bf16x16 a = load_a_f32(xrow + kk + hi * 8);
        bf16x16 b = load_a_bf16(wB + kk + hi * 8);
        acc = wmma_bf16(a, b, acc);
    }
#pragma unroll
    for (int kk = 0; kk < 256; kk += 32) {                    // h @ Us
        bf16x16 a = load_a_bf16(hrow + kk + hi * 8);
        bf16x16 b = load_a_bf16(uB + kk + hi * 8);
        acc = wmma_bf16(a, b, acc);
    }

    int j    = nt * 16 + lo;        // 0..1023 within slice's fused gates
    float bv = biases[s * 1024 + j];
    int gate = j >> 8;              // 0:i 1:f 2:g 3:o (each gate = 16 whole tiles)
    int jj   = j & 255;
    __bf16* out = Gbf + (size_t)gate * (BS * HID) + (size_t)s * H_S + jj;
#pragma unroll
    for (int r = 0; r < 8; ++r) {
        int b_ = mt * 16 + hi * 8 + r;
        float v = acc[r] + bv;
        v = (gate == 2) ? tanhf(v) : sigf(v);
        out[(size_t)b_ * HID] = (__bf16)v;
    }
}

// ---------------------------------------------------------------------------
// Kernel B: connector GEMMs (4 gates) + LSTM cell, fused.
// Block = 256 threads = 8 waves, one 16x16 output tile (mt, nt).
// Wave w: gate g = w/2, K-half kh = w&1 (K=512 each). LDS pairwise reduce.
// ---------------------------------------------------------------------------
__global__ void __launch_bounds__(256)
k_connector(const __bf16* __restrict__ Gbf,   // [4][64][1024] bf16 activated gates
            const __bf16* __restrict__ Ct,    // [4096][1024] bf16 = connector_W^T
            const float* __restrict__ cb,     // [4096] connector bias
            float* __restrict__ c_state,      // [64][1024] fp32
            __bf16* __restrict__ hbf,         // [64][1024] bf16 (next-step A)
            float* __restrict__ hseq,         // [64][512][1024] fp32 output
            float* __restrict__ ht_out,       // [64][1024]
            float* __restrict__ ct_out,       // [64][1024]
            int t, int last) {
    __shared__ float lds[4][256];

    int tile = blockIdx.x;          // 0..255
    int mt = tile >> 6;             // batch tile 0..3
    int nt = tile & 63;             // column tile 0..63
    int wv   = threadIdx.x >> 5;
    int lane = threadIdx.x & 31;
    int g  = wv >> 1;               // gate 0..3
    int kh = wv & 1;                // K-half
    int hi = lane >> 4;
    int lo = lane & 15;

    const __bf16* A = Gbf + (size_t)g * (BS * HID)
                          + (size_t)(mt * 16 + lo) * HID + kh * 512;
    const __bf16* B = Ct + ((size_t)g * 1024 + nt * 16 + lo) * HID + kh * 512;

    v8f acc = {};
#pragma unroll
    for (int kk = 0; kk < 512; kk += 32) {
        bf16x16 a = load_a_bf16(A + kk + hi * 8);
        bf16x16 b = load_a_bf16(B + kk + hi * 8);
        acc = wmma_bf16(a, b, acc);
    }

    if (kh == 1) {
#pragma unroll
        for (int r = 0; r < 8; ++r) lds[g][r * 32 + lane] = acc[r];
    }
    __syncthreads();
    if (kh == 0) {
        float bv = cb[g * 1024 + nt * 16 + lo];
#pragma unroll
        for (int r = 0; r < 8; ++r) {
            float v = acc[r] + lds[g][r * 32 + lane] + bv;
            v = (g == 2) ? tanhf(v) : sigf(v);
            lds[g][r * 32 + lane] = v;      // activated gate value
        }
    }
    __syncthreads();

    // Elementwise LSTM cell: 256 threads, one element each.
    int e  = threadIdx.x;
    int r  = e >> 5;
    int l  = e & 31;
    int b_ = mt * 16 + ((l >> 4) << 3) + r;
    int j  = nt * 16 + (l & 15);
    size_t idx = (size_t)b_ * HID + j;

    float i2 = lds[0][e], f2 = lds[1][e], g2 = lds[2][e], o2 = lds[3][e];
    float c2 = f2 * c_state[idx] + i2 * g2;
    float h2 = o2 * tanhf(c2);
    c_state[idx] = c2;
    hbf[idx]     = (__bf16)h2;
    hseq[((size_t)b_ * TT + t) * HID + j] = h2;
    if (last) { ht_out[idx] = h2; ct_out[idx] = c2; }
}

// ---------------------------------------------------------------------------
// Host launcher
// ---------------------------------------------------------------------------
extern "C" void kernel_launch(void* const* d_in, const int* in_sizes, int n_in,
                              void* d_out, int out_size, void* d_ws, size_t ws_size,
                              hipStream_t stream) {
    const float* x      = (const float*)d_in[0];   // [64,512,1024]
    const float* Ws     = (const float*)d_in[1];   // [4,256,1024]
    const float* Us     = (const float*)d_in[2];   // [4,256,1024]
    const float* biases = (const float*)d_in[3];   // [4,1024]
    const float* cW     = (const float*)d_in[4];   // [1024,4096]
    const float* cb     = (const float*)d_in[5];   // [4096]
    float* out = (float*)d_out;

    // Workspace layout (bf16 counts; everything stays well under ~14 MB)
    __bf16* Wt  = (__bf16*)d_ws;                       // 4*1024*256  = 1 Mi elems
    __bf16* Ut  = Wt + (size_t)NS * 1024 * 256;        // 1 Mi elems
    __bf16* Ct  = Ut + (size_t)NS * 1024 * 256;        // 4096*1024   = 4 Mi elems
    __bf16* hbf = Ct + (size_t)4096 * 1024;            // 64*1024
    __bf16* Gbf = hbf + (size_t)BS * HID;              // 4*64*1024
    float* c_state = (float*)(Gbf + (size_t)4 * BS * HID);

    float* ht_out = out + (size_t)BS * TT * HID;
    float* ct_out = ht_out + (size_t)BS * HID;

    // One-time (per call, deterministic) weight convert+transpose to bf16 [N,K]
    {
        size_t n1 = (size_t)NS * 256 * 1024;
        k_transpose_bf16<<<(unsigned)((n1 + 255) / 256), 256, 0, stream>>>(Ws, Wt, NS, 256, 1024);
        k_transpose_bf16<<<(unsigned)((n1 + 255) / 256), 256, 0, stream>>>(Us, Ut, NS, 256, 1024);
        size_t n2 = (size_t)1024 * 4096;
        k_transpose_bf16<<<(unsigned)((n2 + 255) / 256), 256, 0, stream>>>(cW, Ct, 1, 1024, 4096);
        k_init_state<<<(BS * HID) / 256, 256, 0, stream>>>(hbf, c_state);
    }

    // Recurrence: two WMMA kernels per step; all weights L2-resident after t=0.
    for (int t = 0; t < TT; ++t) {
        k_gates<<<256, 128, 0, stream>>>(x, Wt, Ut, hbf, biases, Gbf, t);
        k_connector<<<256, 256, 0, stream>>>(Gbf, Ct, cb, c_state, hbf,
                                             out, ht_out, ct_out, t, t == TT - 1);
    }
}
